// RCNN_75986561401524
// MI455X (gfx1250) — matrix operand order repaired and checked by reading, not compiled, not run on hardware
//
#include <hip/hip_runtime.h>
#include <cmath>

typedef __bf16 bf16;
typedef __attribute__((ext_vector_type(16))) __bf16 v16bf;
typedef __attribute__((ext_vector_type(8)))  __bf16 v8bf;
typedef __attribute__((ext_vector_type(4)))  __bf16 v4bf;
typedef __attribute__((ext_vector_type(8)))  float  v8f;

#define Bn 512
#define Tn 300
#define Dn 200
#define Hn 128
#define NTT 19   // ceil(300/16) time tiles for the final stage

union V16u { v16bf v; v8bf h[2]; };

__device__ __forceinline__ v8f vzero8() {
  v8f z;
#pragma unroll
  for (int i = 0; i < 8; ++i) z[i] = 0.f;
  return z;
}

// A-matrix fragment (16x32 bf16): lane<16 -> K {kb+0..15}, lane>=16 -> K {kb+16..31}
__device__ __forceinline__ v16bf load_a_frag(const bf16* row, int kbase, int lane) {
  const int koff = (lane & 16) ? 16 : 0;
  return *(const v16bf*)(row + kbase + koff);
}

// B-matrix fragment (32x16 bf16) built from weight row n (row-major (n,k)):
// lane<16 -> K {kb+0..15}, lane>=16 -> K {kb+16..31}; lane&15 selects column n.
__device__ __forceinline__ v16bf load_b_frag(const bf16* wrow, int kbase, int lane) {
  const int koff = (lane & 16) ? 16 : 0;
  return *(const v16bf*)(wrow + kbase + koff);
}

__device__ __forceinline__ v8f wmma_bf(v16bf a, v16bf b, v8f c) {
  return __builtin_amdgcn_wmma_f32_16x16x32_bf16(false, a, false, b, (short)0, c, false, false);
}

// Branch-free fast transcendentals: single v_exp_f32 / v_rcp_f32 each.
// Saturation is numerically correct at the extremes (exp2->inf => rcp->0).
#define LOG2E 1.4426950408889634f
__device__ __forceinline__ float fast_exp(float x) {
  return __builtin_amdgcn_exp2f(x * LOG2E);
}
__device__ __forceinline__ float fast_sig(float x) {
  return __builtin_amdgcn_rcpf(1.f + fast_exp(-x));
}
__device__ __forceinline__ float fast_tanh(float x) {
  return 1.f - 2.f * __builtin_amdgcn_rcpf(fast_exp(2.f * x) + 1.f);
}

// ---------------- weight fp32 -> bf16 (with K zero-padding) ----------------
__global__ void k_convert_pad(const float* __restrict__ src, bf16* __restrict__ dst,
                              int rows, int kin, int kout) {
  int i = blockIdx.x * 256 + threadIdx.x;
  if (i >= rows * kout) return;
  int r = i / kout, c = i - r * kout;
  float v = (c < kin) ? src[r * kin + c] : 0.f;
  dst[i] = (bf16)v;
}

// ---------------- GEMM: out(M,512)bf16 = A(M,K) @ Wb(512,Kp)^T + bias ----------------
// grid.x = M/32 (32-row M tiles), block = 256 (8 waves).
// Wave w owns n-tiles {w, w+8, w+16, w+24}; each B fragment feeds 2 M subtiles.
template <typename SrcT>
__global__ __launch_bounds__(256) void k_gemm512(
    const SrcT* __restrict__ A, int K, int Kp, int permute,
    const bf16* __restrict__ Wb, const float* __restrict__ bias1,
    const float* __restrict__ bias2, bf16* __restrict__ out) {
  __shared__ bf16 sA[32 * 256];
  const int tid = threadIdx.x, lane = tid & 31, wv = tid >> 5;
  const int mbase = blockIdx.x * 32;
  for (int idx = tid; idx < 32 * Kp; idx += 256) {
    int r = idx / Kp, c = idx - r * Kp;
    int gr = mbase + r;
    int arow = permute ? ((gr & (Bn - 1)) * Tn + (gr >> 9)) : gr;  // (t*B+b) -> (b*T+t)
    float v = (c < K) ? (float)A[(size_t)arow * K + c] : 0.f;
    sA[r * Kp + c] = (bf16)v;
  }
  __syncthreads();
  v8f acc[2][4];
#pragma unroll
  for (int m = 0; m < 2; ++m)
#pragma unroll
    for (int u = 0; u < 4; ++u) acc[m][u] = vzero8();
  const int row = lane & 15;
  const int ksteps = Kp >> 5;
  for (int kk = 0; kk < ksteps; ++kk) {
    v16bf aF0 = load_a_frag(&sA[row * Kp], kk * 32, lane);
    v16bf aF1 = load_a_frag(&sA[(16 + row) * Kp], kk * 32, lane);
#pragma unroll
    for (int u = 0; u < 4; ++u) {
      int n = (wv + u * 8) * 16 + row;
      v16bf bF = load_b_frag(Wb + (size_t)n * Kp, kk * 32, lane);
      acc[0][u] = wmma_bf(aF0, bF, acc[0][u]);
      acc[1][u] = wmma_bf(aF1, bF, acc[1][u]);
    }
  }
#pragma unroll
  for (int m = 0; m < 2; ++m) {
    const int rb = mbase + m * 16 + ((lane & 16) ? 8 : 0);
#pragma unroll
    for (int u = 0; u < 4; ++u) {
      int col = (wv + u * 8) * 16 + row;
      float bs = bias1[col] + bias2[col];
#pragma unroll
      for (int j = 0; j < 8; ++j)
        out[(size_t)(rb + j) * 512 + col] = (bf16)(acc[m][u][j] + bs);
    }
  }
}

// ---------------- persistent LSTM recurrence, both directions in one launch ----------------
// grid = (B/16, 2): x = batch tile (16 rows), y = direction (0 fwd, 1 bwd). block = 512 (16 waves).
// gates(16x512) = h(16x128) @ Whh^T(128x512); Whh resident in registers as B-fragments.
__global__ __launch_bounds__(512) void k_lstm(
    const bf16* __restrict__ xpF, const bf16* __restrict__ xpB,  // (T*B,512) rows t*B+b
    const bf16* __restrict__ whhF, const bf16* __restrict__ whhB,
    const float* __restrict__ hf,
    bf16* __restrict__ houtF, bf16* __restrict__ houtB,  // rows t*B+b, stride hstride
    int hstride, int hoffF, int hoffB) {
  __shared__ bf16 sH[16 * 128];
  __shared__ float sG[16 * 512];
  const int dir = blockIdx.y;
  const bf16* xp = dir ? xpB : xpF;
  const bf16* whh = dir ? whhB : whhF;
  bf16* hout = dir ? houtB : houtF;
  const int hoff = dir ? hoffB : hoffF;

  const int tid = threadIdx.x, lane = tid & 31, wv = tid >> 5;
  const int bbase = blockIdx.x * 16;
  const int row = lane & 15;

  // preload Whh fragments: wave wv owns n-tiles {wv, wv+16}
  v16bf bW[2][4];
#pragma unroll
  for (int p = 0; p < 2; ++p) {
    int n = (wv + p * 16) * 16 + row;
#pragma unroll
    for (int kk = 0; kk < 4; ++kk)
      bW[p][kk] = load_b_frag(whh + (size_t)n * 128, kk * 32, lane);
  }

  // state: thread owns (batch row = wv, hidden j0..j0+3)
  const int j0 = lane * 4;
  const int brow = wv;
  float cR[4];
#pragma unroll
  for (int u = 0; u < 4; ++u) {
    float v = hf[j0 + u];
    cR[u] = v;
    sH[brow * 128 + j0 + u] = (bf16)v;
  }
  __syncthreads();

  for (int s = 0; s < Tn; ++s) {
    const int t = dir ? (Tn - 1 - s) : s;
    // ---- issue xp loads early: independent of LDS, overlap with WMMA phase ----
    const size_t xrow = ((size_t)t * Bn + bbase + brow) * 512;
    v4bf xi = *(const v4bf*)(xp + xrow + 0 + j0);
    v4bf xf = *(const v4bf*)(xp + xrow + 128 + j0);
    v4bf xc = *(const v4bf*)(xp + xrow + 256 + j0);
    v4bf xo = *(const v4bf*)(xp + xrow + 384 + j0);
    // ---- matmul phase ----
    v8f acc0 = vzero8(), acc1 = vzero8();
#pragma unroll
    for (int kk = 0; kk < 4; ++kk) {
      v16bf aF = load_a_frag(&sH[row * 128], kk * 32, lane);
      acc0 = wmma_bf(aF, bW[0][kk], acc0);
      acc1 = wmma_bf(aF, bW[1][kk], acc1);
    }
    const int r0 = (lane & 16) ? 8 : 0;
    {
      int c0 = wv * 16 + row, c1 = (wv + 16) * 16 + row;
#pragma unroll
      for (int j = 0; j < 8; ++j) {
        sG[(r0 + j) * 512 + c0] = acc0[j];
        sG[(r0 + j) * 512 + c1] = acc1[j];
      }
    }
    __syncthreads();
    // ---- activation phase (branch-free v_exp/v_rcp) ----
    const float* pg = &sG[brow * 512];
    v4bf hh;
#pragma unroll
    for (int u = 0; u < 4; ++u) {
      float gI = pg[0 + j0 + u] + (float)xi[u];
      float gF = pg[128 + j0 + u] + (float)xf[u];
      float gC = pg[256 + j0 + u] + (float)xc[u];
      float gO = pg[384 + j0 + u] + (float)xo[u];
      cR[u] = fast_sig(gF) * cR[u] + fast_sig(gI) * fast_tanh(gC);
      float h = fast_sig(gO) * fast_tanh(cR[u]);
      sH[brow * 128 + j0 + u] = (bf16)h;
      hh[u] = (bf16)h;
    }
    *(v4bf*)(hout + ((size_t)t * Bn + bbase + brow) * hstride + hoff + j0) = hh;
    __syncthreads();
  }
}

// ---------------- final: assemble [mat_FW | b_x | mat_BW] tile, GEMM, tanh, tile-max ----------------
// grid = B * NTT, block = 256 (8 waves, wave = n-tile of 128 cols)
__global__ __launch_bounds__(256) void k_final(
    const bf16* __restrict__ fw, const bf16* __restrict__ bw,
    const float* __restrict__ bx, const float* __restrict__ hf,
    const bf16* __restrict__ wl1,  // (128, 480) bf16, cols 456..479 zero
    float* __restrict__ pmax) {    // (B, NTT, 128)
  __shared__ bf16 sA[16 * 480];
  const int tid = threadIdx.x, lane = tid & 31, wv = tid >> 5;
  const int b = blockIdx.x / NTT, tt = blockIdx.x - b * NTT;
  const int t0 = tt * 16;
  for (int idx = tid; idx < 16 * 480; idx += 256) {
    int r = idx / 480, c = idx - r * 480;
    int t = t0 + r;
    float v = 0.f;
    if (t < Tn && c < 456) {
      if (c < 128)
        v = (t == 0) ? hf[c] : (float)fw[((size_t)t * Bn + b) * 128 + c];
      else if (c < 328)
        v = bx[((size_t)b * Tn + t) * Dn + (c - 128)];
      else {
        int j = c - 328;
        v = (t == Tn - 1) ? hf[j] : (float)bw[((size_t)(Tn - 2 - t) * Bn + b) * 128 + j];
      }
    }
    sA[idx] = (bf16)v;
  }
  __syncthreads();
  v8f acc = vzero8();
  const int row = lane & 15;
  for (int kk = 0; kk < 15; ++kk) {
    v16bf aF = load_a_frag(&sA[row * 480], kk * 32, lane);
    v16bf bF = load_b_frag(wl1 + (size_t)(wv * 16 + row) * 480, kk * 32, lane);
    acc = wmma_bf(aF, bF, acc);
  }
  const int r0 = (lane & 16) ? 8 : 0;
  float m = -1e30f;
#pragma unroll
  for (int j = 0; j < 8; ++j) {
    int t = t0 + r0 + j;
    if (t < Tn) m = fmaxf(m, fast_tanh(acc[j]));
  }
  m = fmaxf(m, __shfl_xor(m, 16, 32));
  if (lane < 16) pmax[((size_t)b * NTT + tt) * 128 + wv * 16 + lane] = m;
}

// ---------------- reduce over time tiles + classifier + b_y passthrough ----------------
__global__ void k_reduce_cls(const float* __restrict__ pmax, const float* __restrict__ w_l2,
                             const float* __restrict__ b_l2, const int* __restrict__ b_y,
                             float* __restrict__ out, int ocap) {
  __shared__ float sp[128];
  const int b = blockIdx.x, j = threadIdx.x;
  float m = -1e30f;
  for (int tt = 0; tt < NTT; ++tt) m = fmaxf(m, pmax[((size_t)b * NTT + tt) * 128 + j]);
  sp[j] = m;
  __syncthreads();
  if (j < 4) {
    float s = b_l2[j];
    for (int k = 0; k < 128; ++k) s += sp[k] * w_l2[j * 128 + k];
    out[b * 4 + j] = s;
  }
  if (j == 4 && (Bn * 4 + b) < ocap) ((int*)out)[Bn * 4 + b] = b_y[b];
}

extern "C" void kernel_launch(void* const* d_in, const int* in_sizes, int n_in,
                              void* d_out, int out_size, void* d_ws, size_t ws_size,
                              hipStream_t stream) {
  const float* bx = (const float*)d_in[0];
  const int*   by = (const int*)d_in[1];
  const float* hf = (const float*)d_in[2];
  const float* w_ih[4] = {(const float*)d_in[3], (const float*)d_in[7],
                          (const float*)d_in[11], (const float*)d_in[15]};
  const float* w_hh[4] = {(const float*)d_in[4], (const float*)d_in[8],
                          (const float*)d_in[12], (const float*)d_in[16]};
  const float* b_ih[4] = {(const float*)d_in[5], (const float*)d_in[9],
                          (const float*)d_in[13], (const float*)d_in[17]};
  const float* b_hh[4] = {(const float*)d_in[6], (const float*)d_in[10],
                          (const float*)d_in[14], (const float*)d_in[18]};
  const float* w_l1 = (const float*)d_in[19];
  const float* w_l2 = (const float*)d_in[20];
  const float* b_l2 = (const float*)d_in[21];

  char* wsb = (char*)d_ws;
  size_t off = 0;
  auto alloc = [&](size_t bytes) {
    size_t o = off;
    off = (off + bytes + 255) & ~(size_t)255;
    return o;
  };
  const size_t o_wih0f = alloc((size_t)512 * 224 * 2);
  const size_t o_wih0b = alloc((size_t)512 * 224 * 2);
  const size_t o_wih1f = alloc((size_t)512 * 256 * 2);
  const size_t o_wih1b = alloc((size_t)512 * 256 * 2);
  const size_t o_whh[4] = {alloc((size_t)512 * 128 * 2), alloc((size_t)512 * 128 * 2),
                           alloc((size_t)512 * 128 * 2), alloc((size_t)512 * 128 * 2)};
  const size_t o_wl1 = alloc((size_t)128 * 480 * 2);
  const size_t o_xpA = alloc((size_t)Tn * Bn * 512 * 2);
  const size_t o_xpB = alloc((size_t)Tn * Bn * 512 * 2);
  const size_t o_y0  = alloc((size_t)Tn * Bn * 256 * 2);
  const size_t o_fw  = alloc((size_t)Tn * Bn * 128 * 2);
  const size_t o_bw  = alloc((size_t)Tn * Bn * 128 * 2);
  const size_t o_pm  = alloc((size_t)Bn * NTT * 128 * 4);
  (void)ws_size;

  bf16* wih0f = (bf16*)(wsb + o_wih0f);
  bf16* wih0b = (bf16*)(wsb + o_wih0b);
  bf16* wih1f = (bf16*)(wsb + o_wih1f);
  bf16* wih1b = (bf16*)(wsb + o_wih1b);
  bf16* whhb[4];
  for (int i = 0; i < 4; ++i) whhb[i] = (bf16*)(wsb + o_whh[i]);
  bf16* wl1b = (bf16*)(wsb + o_wl1);
  bf16* xpA = (bf16*)(wsb + o_xpA);
  bf16* xpB = (bf16*)(wsb + o_xpB);
  bf16* y0  = (bf16*)(wsb + o_y0);
  bf16* fwb = (bf16*)(wsb + o_fw);
  bf16* bwb = (bf16*)(wsb + o_bw);
  float* pm = (float*)(wsb + o_pm);

  auto cvt = [&](const float* s, bf16* d, int rows, int kin, int kout) {
    int tot = rows * kout;
    k_convert_pad<<<(tot + 255) / 256, 256, 0, stream>>>(s, d, rows, kin, kout);
  };
  cvt(w_ih[0], wih0f, 512, 200, 224);
  cvt(w_ih[1], wih0b, 512, 200, 224);
  cvt(w_ih[2], wih1f, 512, 256, 256);
  cvt(w_ih[3], wih1b, 512, 256, 256);
  for (int i = 0; i < 4; ++i) cvt(w_hh[i], whhb[i], 512, 128, 128);
  cvt(w_l1, wl1b, 128, 456, 480);

  const int Mblk = (Tn * Bn) / 32;  // 4800
  // layer 0 input projections (A = b_x fp32, row-permuted (b,t) -> (t,b))
  k_gemm512<float><<<Mblk, 256, 0, stream>>>(bx, 200, 224, 1, wih0f, b_ih[0], b_hh[0], xpA);
  k_gemm512<float><<<Mblk, 256, 0, stream>>>(bx, 200, 224, 1, wih0b, b_ih[1], b_hh[1], xpB);
  // layer 0 recurrences, both directions concurrently -> y0 (cols 0..127 fwd, 128..255 bwd)
  k_lstm<<<dim3(Bn / 16, 2), 512, 0, stream>>>(xpA, xpB, whhb[0], whhb[1], hf, y0, y0, 256, 0, 128);
  // layer 1 input projections (A = y0 bf16, already (t,b)-ordered)
  k_gemm512<bf16><<<Mblk, 256, 0, stream>>>(y0, 256, 256, 0, wih1f, b_ih[2], b_hh[2], xpA);
  k_gemm512<bf16><<<Mblk, 256, 0, stream>>>(y0, 256, 256, 0, wih1b, b_ih[3], b_hh[3], xpB);
  // layer 1 recurrences, both directions concurrently
  k_lstm<<<dim3(Bn / 16, 2), 512, 0, stream>>>(xpA, xpB, whhb[2], whhb[3], hf, fwb, bwb, 128, 0, 0);
  // final linear + tanh + tile max, then reduce + classifier
  k_final<<<Bn * NTT, 256, 0, stream>>>(fwb, bwb, bx, hf, wl1b, pm);
  k_reduce_cls<<<Bn, 128, 0, stream>>>(pm, w_l2, b_l2, by, (float*)d_out, out_size);
}
